// RNN_52982716564334
// MI455X (gfx1250) — matrix operand order repaired and checked by reading, not compiled, not run on hardware
//
#include <hip/hip_runtime.h>
#include <hip/hip_bf16.h>

// MI455X / gfx1250 LSTM stack, v5 (round-3 structure + native (__bf16) converts):
//   xprep  : X -> bf16 WMMA A-fragment tiles
//   xgemm1 : Xf @ W1_ih^T + b1 -> gate tiles (D-layout bf16)   [parallel, WMMA]
//   lstm1  : recurrence only (h @ W1_hh), hh weights in VGPRs (128),
//            depth-2 async tile prefetch (s_wait_asynccnt 8), writes hs1f
//   xgemm2 : hs1f @ W2_ih^T + b2 -> gate tiles                 [parallel, WMMA]
//   lstm2  : recurrence only (h @ W2_hh), 8 WMMA/step/wave -> h_n
//   head   : FC 64->32 relu -> 1 sigmoid
// Workspace layout (~768 MB):
//   [0,128M)    Xf   (dead after xgemm1)  / XW2 (256M, written after lstm1)
//   [128M,640M) XW1  (dead after lstm1)
//   [640M,768M) hs1f
//   [768M,..)   hn (256 KB)

typedef __attribute__((ext_vector_type(16))) __bf16 v16bf;
typedef __attribute__((ext_vector_type(8)))  float  v8f;

#define B_SZ 1024
#define T_SZ 512
#define EMB  100
#define H1   128
#define H2   64
#define BT   32
#define TILE_BYTES 8192                          // A-frag tile: 8 frags * 1KB
#define NBLK (B_SZ / BT)                         // 32
#define NTILES ((size_t)NBLK * T_SZ)             // 16384
#define XF_BYTES (NTILES * TILE_BYTES)           // 128 MB
#define XW1_TILE 32768                           // 32 nt * 2 mt * 512B
#define XW2_TILE 16384                           // 16 nt * 2 mt * 512B

// f32 -> bf16 via the compiler's native fptrunc (hardware cvt on gfx1250)
__device__ __forceinline__ unsigned short f2bf(float x) {
  return __builtin_bit_cast(unsigned short, (__bf16)x);
}

// A-matrix (16x32 bf16) fragment byte offset for element (row m, col kk), contiguous K per lane half (k = 16h + s)
__device__ __forceinline__ int a_off(int m, int kk) {
  int h = kk >> 4, j = (kk & 15) >> 1;
  return ((m & 15) + (h << 4)) * 32 + j * 4 + (kk & 1) * 2;
}

__device__ __forceinline__ float sigm(float x) {
  return __builtin_amdgcn_rcpf(1.f + __expf(-x));
}
__device__ __forceinline__ float tanh_fast(float x) {
  float e = __expf(2.f * x);
  return 1.f - 2.f * __builtin_amdgcn_rcpf(e + 1.f);
}

// ---- CDNA5 async global->LDS (ASYNCcnt; loads only => in-order completion) ----
__device__ __forceinline__ void async_ld16(unsigned lds_off, const void* g) {
  asm volatile("global_load_async_to_lds_b128 %0, %1, off"
               :: "v"(lds_off), "v"(g) : "memory");
}
__device__ __forceinline__ void wait_async_le8() { asm volatile("s_wait_asynccnt 0x8" ::: "memory"); }
__device__ __forceinline__ void wait_async_le4() { asm volatile("s_wait_asynccnt 0x4" ::: "memory"); }
__device__ __forceinline__ void wait_async0()    { asm volatile("s_wait_asynccnt 0x0" ::: "memory"); }

// B-matrix (32x16 bf16) fragment loaded straight from global into VGPRs.
// Lane L holds column n=(L&15), rows kbase..kbase+15 (contiguous K per lane).
__device__ __forceinline__ v16bf load_bfrag(const float* __restrict__ W, int ldw,
                                            int n, int kbase, int kvalid) {
  v16bf v;
#pragma unroll
  for (int e = 0; e < 16; ++e) {
    int k = kbase + e;
    float x = (k < kvalid) ? W[n * ldw + k] : 0.f;
    v[e] = (__bf16)x;
  }
  return v;
}

// D-subtile (512B): lane L holds its 8 C/D values contiguously at [L*16, L*16+16)
__device__ __forceinline__ uint4 pack_bf8(v8f a) {
  union { uint4 u; __bf16 b[8]; } o;
#pragma unroll
  for (int r = 0; r < 8; ++r) o.b[r] = (__bf16)a[r];
  return o.u;
}
__device__ __forceinline__ v8f unpack_bf8(const void* p) {
  uint4 d = *(const uint4*)p;
  v8f r;
  r[0] = __builtin_bit_cast(float, d.x << 16);
  r[1] = __builtin_bit_cast(float, d.x & 0xffff0000u);
  r[2] = __builtin_bit_cast(float, d.y << 16);
  r[3] = __builtin_bit_cast(float, d.y & 0xffff0000u);
  r[4] = __builtin_bit_cast(float, d.z << 16);
  r[5] = __builtin_bit_cast(float, d.z & 0xffff0000u);
  r[6] = __builtin_bit_cast(float, d.w << 16);
  r[7] = __builtin_bit_cast(float, d.w & 0xffff0000u);
  return r;
}

#define WMMA_BF16(A, B, C) \
  __builtin_amdgcn_wmma_f32_16x16x32_bf16(false, (A), false, (B), (short)0, (C), false, false)

// ------------------------------------------------------------------ xprep ----
__global__ __launch_bounds__(256) void xprep_kernel(const float* __restrict__ X,
                                                    unsigned short* __restrict__ Xf) {
  __shared__ unsigned char lds[TILE_BYTES];
  const int tid = threadIdx.x;
  const int blk = blockIdx.x >> 9;
  const int t   = blockIdx.x & (T_SZ - 1);
  const int b0  = blk * BT;
  for (int i = tid; i < BT * 128; i += 256) {
    int m = i >> 7, k = i & 127;
    float v = (k < EMB) ? X[((size_t)(b0 + m) * T_SZ + t) * EMB + k] : 0.f;
    int off = ((k >> 5) * 2 + (m >> 4)) * 1024 + a_off(m, k & 31);
    *(unsigned short*)(lds + off) = f2bf(v);
  }
  __syncthreads();
  const uint4* s = (const uint4*)lds;
  uint4* d = (uint4*)((char*)Xf + ((size_t)blk * T_SZ + t) * TILE_BYTES);
#pragma unroll
  for (int q = 0; q < 2; ++q) d[tid * 2 + q] = s[tid * 2 + q];
}

// ------------------------------------------------------------------ xgemm1 ---
// Per (blk,t): [32x128 bf16] @ [128x512] + bias -> 32KB D-tile. No LDS at all.
__global__ __launch_bounds__(256) void xgemm1_kernel(
    const unsigned short* __restrict__ Xf, const float* __restrict__ Wih,
    const float* __restrict__ bih, const float* __restrict__ bhh,
    unsigned short* __restrict__ XW1) {
  const int tid = threadIdx.x, lane = tid & 31, w = tid >> 5;
  const int half = lane >> 4;
  const size_t tile = blockIdx.x;
  const char* atile = (const char*)Xf + tile * (size_t)TILE_BYTES;

  v16bf wfr[4][4];
  float bias[4];
  const int ncol = (w << 4) + (lane & 15);
#pragma unroll
  for (int g = 0; g < 4; ++g) {
    int n = g * H1 + ncol;
    bias[g] = bih[n] + bhh[n];
#pragma unroll
    for (int kt = 0; kt < 4; ++kt)
      wfr[kt][g] = load_bfrag(Wih, EMB, n, kt * 32 + half * 16, EMB);
  }

  v8f acc[4][2];
#pragma unroll
  for (int g = 0; g < 4; ++g)
#pragma unroll
    for (int mt = 0; mt < 2; ++mt)
#pragma unroll
      for (int r = 0; r < 8; ++r) acc[g][mt][r] = bias[g];

#pragma unroll
  for (int kt = 0; kt < 4; ++kt) {
    v16bf a0 = *(const v16bf*)(atile + (kt * 2 + 0) * 1024 + lane * 32);
    v16bf a1 = *(const v16bf*)(atile + (kt * 2 + 1) * 1024 + lane * 32);
#pragma unroll
    for (int g = 0; g < 4; ++g) {
      acc[g][0] = WMMA_BF16(a0, wfr[kt][g], acc[g][0]);
      acc[g][1] = WMMA_BF16(a1, wfr[kt][g], acc[g][1]);
    }
  }

  char* dtile = (char*)XW1 + tile * (size_t)XW1_TILE;
#pragma unroll
  for (int g = 0; g < 4; ++g)
#pragma unroll
    for (int mt = 0; mt < 2; ++mt)
      *(uint4*)(dtile + ((g * 8 + w) * 2 + mt) * 512 + lane * 16) = pack_bf8(acc[g][mt]);
}

// ------------------------------------------------------------------ xgemm2 ---
__global__ __launch_bounds__(256) void xgemm2_kernel(
    const unsigned short* __restrict__ hs1f, const float* __restrict__ Wih,
    const float* __restrict__ bih, const float* __restrict__ bhh,
    unsigned short* __restrict__ XW2) {
  const int tid = threadIdx.x, lane = tid & 31, w = tid >> 5;
  const int half = lane >> 4;
  const size_t tile = blockIdx.x;
  const char* atile = (const char*)hs1f + tile * (size_t)TILE_BYTES;

  v16bf wfr[4][2];
  float bias[2];
#pragma unroll
  for (int q = 0; q < 2; ++q) {
    int n = (q * 8 + w) * 16 + (lane & 15);
    bias[q] = bih[n] + bhh[n];
#pragma unroll
    for (int kt = 0; kt < 4; ++kt)
      wfr[kt][q] = load_bfrag(Wih, H1, n, kt * 32 + half * 16, H1);
  }

  v8f acc[2][2];
#pragma unroll
  for (int q = 0; q < 2; ++q)
#pragma unroll
    for (int mt = 0; mt < 2; ++mt)
#pragma unroll
      for (int r = 0; r < 8; ++r) acc[q][mt][r] = bias[q];

#pragma unroll
  for (int kt = 0; kt < 4; ++kt) {
    v16bf a0 = *(const v16bf*)(atile + (kt * 2 + 0) * 1024 + lane * 32);
    v16bf a1 = *(const v16bf*)(atile + (kt * 2 + 1) * 1024 + lane * 32);
#pragma unroll
    for (int q = 0; q < 2; ++q) {
      acc[q][0] = WMMA_BF16(a0, wfr[kt][q], acc[q][0]);
      acc[q][1] = WMMA_BF16(a1, wfr[kt][q], acc[q][1]);
    }
  }

  char* dtile = (char*)XW2 + tile * (size_t)XW2_TILE;
#pragma unroll
  for (int q = 0; q < 2; ++q)
#pragma unroll
    for (int mt = 0; mt < 2; ++mt)
      *(uint4*)(dtile + ((q * 8 + w) * 2 + mt) * 512 + lane * 16) = pack_bf8(acc[q][mt]);
}

// ------------------------------------------------------------------ lstm1 ----
// LDS: 3 x 32KB XW tile ring | 8KB h fragments = 104KB
#define HF1 98304u
#define LDS1_BYTES (98304 + 8192)

__global__ __launch_bounds__(256) void lstm1_kernel(
    const unsigned short* __restrict__ XW1, const float* __restrict__ Whh,
    unsigned short* __restrict__ hs1f) {
  extern __shared__ unsigned char lds[];
  const int tid = threadIdx.x, lane = tid & 31, w = tid >> 5;
  const int half = lane >> 4;
  const size_t blk = blockIdx.x;
  const unsigned bufs[3] = {0u, 32768u, 65536u};

  // prologue: prefetch gate tiles t=0,1 (depth-2 pipeline)
#pragma unroll
  for (int pt = 0; pt < 2; ++pt) {
    const char* src = (const char*)XW1 + (blk * T_SZ + pt) * (size_t)XW1_TILE + tid * 128;
#pragma unroll
    for (int q = 0; q < 8; ++q) async_ld16(bufs[pt] + tid * 128 + q * 16, src + q * 16);
  }

  // 16 hh B-fragments resident in VGPRs (128 VGPRs)
  v16bf wf[4][4];
  const int ncol = (w << 4) + (lane & 15);
#pragma unroll
  for (int kt = 0; kt < 4; ++kt)
#pragma unroll
    for (int g = 0; g < 4; ++g)
      wf[kt][g] = load_bfrag(Whh, H1, g * H1 + ncol, kt * 32 + half * 16, H1);

  for (int i = tid; i < 2048; i += 256)          // h0 = 0
    ((unsigned int*)(lds + HF1))[i] = 0u;

  v8f cst[2];
#pragma unroll
  for (int mt = 0; mt < 2; ++mt)
#pragma unroll
    for (int r = 0; r < 8; ++r) cst[mt][r] = 0.f;

  const int ktp = ncol >> 5, kkp = ncol & 31;
  int cur = 0;

  for (int t = 0; t < T_SZ; ++t) {
    if (t + 1 < T_SZ) wait_async_le8(); else wait_async0();  // tile t ready (in-order)
    __syncthreads();

    if (t + 2 < T_SZ) {
      int nb = cur + 2; if (nb >= 3) nb -= 3;
      const char* src = (const char*)XW1 + (blk * T_SZ + t + 2) * (size_t)XW1_TILE + tid * 128;
#pragma unroll
      for (int q = 0; q < 8; ++q) async_ld16(bufs[nb] + tid * 128 + q * 16, src + q * 16);
    }

    const unsigned xb = bufs[cur];
    v8f acc[4][2];
#pragma unroll
    for (int g = 0; g < 4; ++g)
#pragma unroll
      for (int mt = 0; mt < 2; ++mt)
        acc[g][mt] = unpack_bf8(lds + xb + ((g * 8 + w) * 2 + mt) * 512 + lane * 16);

#pragma unroll
    for (int kt = 0; kt < 4; ++kt) {
      v16bf a0 = *(const v16bf*)(lds + HF1 + kt * 2048 + lane * 32);
      v16bf a1 = *(const v16bf*)(lds + HF1 + kt * 2048 + 1024 + lane * 32);
#pragma unroll
      for (int g = 0; g < 4; ++g) {
        acc[g][0] = WMMA_BF16(a0, wf[kt][g], acc[g][0]);
        acc[g][1] = WMMA_BF16(a1, wf[kt][g], acc[g][1]);
      }
    }
    __syncthreads();        // all h reads done before overwrite

    char* dtile = (char*)hs1f + (blk * T_SZ + t) * (size_t)TILE_BYTES;
#pragma unroll
    for (int mt = 0; mt < 2; ++mt) {
#pragma unroll
      for (int r = 0; r < 8; ++r) {
        float iv = acc[0][mt][r], fv = acc[1][mt][r];
        float gv = acc[2][mt][r], ov = acc[3][mt][r];
        float cn = sigm(fv) * cst[mt][r] + sigm(iv) * tanh_fast(gv);
        cst[mt][r] = cn;
        float hv = sigm(ov) * tanh_fast(cn);
        unsigned short hb = f2bf(hv);
        int m = mt * 16 + r + half * 8;
        int fo = (ktp * 2 + mt) * 1024 + a_off(m, kkp);
        *(unsigned short*)(lds + HF1 + fo) = hb;   // recurrence state
        *(unsigned short*)(dtile + fo) = hb;       // hs1f (fire-and-forget)
      }
    }
    cur = cur + 1; if (cur == 3) cur = 0;
  }
}

// ------------------------------------------------------------------ lstm2 ----
// LDS: 3 x 16KB XW2 tile ring | 4KB h2 fragments = 52KB
#define HF2 49152u
#define LDS2_BYTES (49152 + 4096)

__global__ __launch_bounds__(256) void lstm2_kernel(
    const unsigned short* __restrict__ XW2, const float* __restrict__ Whh,
    float* __restrict__ hn) {
  extern __shared__ unsigned char lds[];
  const int tid = threadIdx.x, lane = tid & 31, w = tid >> 5;
  const int half = lane >> 4;
  const int mq = w >> 2, wl = w & 3;
  const size_t blk = blockIdx.x;
  const unsigned bufs[3] = {0u, 16384u, 32768u};

#pragma unroll
  for (int pt = 0; pt < 2; ++pt) {
    const char* src = (const char*)XW2 + (blk * T_SZ + pt) * (size_t)XW2_TILE + tid * 64;
#pragma unroll
    for (int q = 0; q < 4; ++q) async_ld16(bufs[pt] + tid * 64 + q * 16, src + q * 16);
  }

  v16bf wf[2][4];
  const int ncol = (wl << 4) + (lane & 15);
#pragma unroll
  for (int kt = 0; kt < 2; ++kt)
#pragma unroll
    for (int g = 0; g < 4; ++g)
      wf[kt][g] = load_bfrag(Whh, H2, g * H2 + ncol, kt * 32 + half * 16, H2);

  for (int i = tid; i < 1024; i += 256)          // h0 = 0
    ((unsigned int*)(lds + HF2))[i] = 0u;

  v8f cst;
#pragma unroll
  for (int r = 0; r < 8; ++r) cst[r] = 0.f;

  const int ktp = ncol >> 5, kkp = ncol & 31;
  int cur = 0;

  for (int t = 0; t < T_SZ; ++t) {
    if (t + 1 < T_SZ) wait_async_le4(); else wait_async0();
    __syncthreads();

    if (t + 2 < T_SZ) {
      int nb = cur + 2; if (nb >= 3) nb -= 3;
      const char* src = (const char*)XW2 + (blk * T_SZ + t + 2) * (size_t)XW2_TILE + tid * 64;
#pragma unroll
      for (int q = 0; q < 4; ++q) async_ld16(bufs[nb] + tid * 64 + q * 16, src + q * 16);
    }

    const unsigned xb = bufs[cur];
    v8f acc[4];
#pragma unroll
    for (int g = 0; g < 4; ++g)
      acc[g] = unpack_bf8(lds + xb + ((g * 4 + wl) * 2 + mq) * 512 + lane * 16);

#pragma unroll
    for (int kt = 0; kt < 2; ++kt) {
      v16bf a = *(const v16bf*)(lds + HF2 + (kt * 2 + mq) * 1024 + lane * 32);
#pragma unroll
      for (int g = 0; g < 4; ++g)
        acc[g] = WMMA_BF16(a, wf[kt][g], acc[g]);
    }
    __syncthreads();

#pragma unroll
    for (int r = 0; r < 8; ++r) {
      float iv = acc[0][r], fv = acc[1][r], gv = acc[2][r], ov = acc[3][r];
      float cn = sigm(fv) * cst[r] + sigm(iv) * tanh_fast(gv);
      cst[r] = cn;
      float hv = sigm(ov) * tanh_fast(cn);
      int m = r + half * 8;
      *(unsigned short*)(lds + HF2 + (ktp * 2 + mq) * 1024 + a_off(m, kkp)) = f2bf(hv);
      if (t == T_SZ - 1)
        hn[(size_t)(blk * BT + mq * 16 + m) * H2 + ncol] = hv;
    }
    cur = cur + 1; if (cur == 3) cur = 0;
  }
}

// ------------------------------------------------------------------ FC head --
__global__ __launch_bounds__(256) void head_kernel(
    const float* __restrict__ hn, const float* __restrict__ fc1_w,
    const float* __restrict__ fc1_b, const float* __restrict__ fc2_w,
    const float* __restrict__ fc2_b, float* __restrict__ out) {
  int b = blockIdx.x * blockDim.x + threadIdx.x;
  if (b >= B_SZ) return;
  float h[H2];
#pragma unroll
  for (int j = 0; j < H2; ++j) h[j] = hn[(size_t)b * H2 + j];
  float acc2 = fc2_b[0];
  for (int o = 0; o < 32; ++o) {
    float s = fc1_b[o];
#pragma unroll
    for (int j = 0; j < H2; ++j) s += fc1_w[o * H2 + j] * h[j];
    s = fmaxf(s, 0.f);
    acc2 += fc2_w[o] * s;
  }
  out[b] = __builtin_amdgcn_rcpf(1.f + __expf(-acc2));
}

extern "C" void kernel_launch(void* const* d_in, const int* in_sizes, int n_in,
                              void* d_out, int out_size, void* d_ws, size_t ws_size,
                              hipStream_t stream) {
  (void)in_sizes; (void)n_in; (void)out_size; (void)ws_size;
  const float* X     = (const float*)d_in[0];
  const float* W1_ih = (const float*)d_in[1];
  const float* W1_hh = (const float*)d_in[2];
  const float* b1_ih = (const float*)d_in[3];
  const float* b1_hh = (const float*)d_in[4];
  const float* W2_ih = (const float*)d_in[5];
  const float* W2_hh = (const float*)d_in[6];
  const float* b2_ih = (const float*)d_in[7];
  const float* b2_hh = (const float*)d_in[8];
  const float* fc1_w = (const float*)d_in[9];
  const float* fc1_b = (const float*)d_in[10];
  const float* fc2_w = (const float*)d_in[11];
  const float* fc2_b = (const float*)d_in[12];

  char* ws = (char*)d_ws;
  unsigned short* Xf   = (unsigned short*)ws;                                  // 128 MB
  unsigned short* XW1  = (unsigned short*)(ws + XF_BYTES);                     // 512 MB
  unsigned short* hs1f = (unsigned short*)(ws + XF_BYTES + NTILES * XW1_TILE); // 128 MB
  unsigned short* XW2  = (unsigned short*)ws;          // reuses Xf/XW1 (dead by then)
  float*          hn   = (float*)(ws + 2 * XF_BYTES + NTILES * XW1_TILE);      // 256 KB

  xprep_kernel <<<NTILES, 256, 0, stream>>>(X, Xf);
  xgemm1_kernel<<<NTILES, 256, 0, stream>>>(Xf, W1_ih, b1_ih, b1_hh, XW1);
  lstm1_kernel <<<NBLK, 256, LDS1_BYTES, stream>>>(XW1, W1_hh, hs1f);
  xgemm2_kernel<<<NTILES, 256, 0, stream>>>(hs1f, W2_ih, b2_ih, b2_hh, XW2);
  lstm2_kernel <<<NBLK, 256, LDS2_BYTES, stream>>>(XW2, W2_hh, hn);
  head_kernel  <<<B_SZ / 256, 256, 0, stream>>>(hn, fc1_w, fc1_b, fc2_w, fc2_b, (float*)d_out);
}
